// MambaBlockWrapper_60138132078599
// MI455X (gfx1250) — compile-verified, matrix operand order checked by reading, not run on hardware
//
#include <hip/hip_runtime.h>

#define B_ 4
#define T_ 2048
#define D_ 1024
#define N_ 16
#define M_ (B_ * T_)   // 8192 rows

#define TILE_M 128
#define TILE_N 64
#define TILE_K 32
#define LDSS   40      // ushorts per LDS row (32 data + 8 pad) -> 80B rows, 16B aligned

typedef __attribute__((ext_vector_type(16))) __bf16       v16bf;
typedef __attribute__((ext_vector_type(8)))  float        v8f;
typedef __attribute__((ext_vector_type(4)))  unsigned int uint4v;
typedef __attribute__((ext_vector_type(2)))  unsigned int uint2v;

struct Frag { union { v16bf v; uint4v q[2]; }; };

__device__ __forceinline__ unsigned short f2bf(float f) {
  unsigned int u = __float_as_uint(f);
  unsigned int r = u + 0x7FFFu + ((u >> 16) & 1u);   // round-to-nearest-even
  return (unsigned short)(r >> 16);
}
__device__ __forceinline__ float bf2f(unsigned short h) {
  return __uint_as_float(((unsigned int)h) << 16);
}
__device__ __forceinline__ float softplusf(float v) {
  return fmaxf(v, 0.0f) + log1pf(__expf(-fabsf(v)));
}

// Async DMA: 16 bytes/lane, global -> LDS, tracked by ASYNCcnt (CDNA5).
__device__ __forceinline__ void async_ld16(const unsigned short* ldsdst,
                                           const void* gbase, unsigned gvoff) {
  unsigned ldsoff = (unsigned)(size_t)ldsdst;   // low 32 bits = LDS byte offset
  asm volatile("global_load_async_to_lds_b128 %0, %1, %2"
               :: "v"(ldsoff), "v"(gvoff), "s"(gbase)
               : "memory");
}

// Split fp32 array into bf16 hi/lo arrays (one pass, done once per operand).
__global__ __launch_bounds__(256) void split_kernel(
    const float* __restrict__ src, unsigned short* __restrict__ hi,
    unsigned short* __restrict__ lo, int n4) {
  const int i = blockIdx.x * blockDim.x + threadIdx.x;
  if (i >= n4) return;
  const float4 v = ((const float4*)src)[i];
  const unsigned short h0 = f2bf(v.x), h1 = f2bf(v.y), h2 = f2bf(v.z), h3 = f2bf(v.w);
  const unsigned short l0 = f2bf(v.x - bf2f(h0));
  const unsigned short l1 = f2bf(v.y - bf2f(h1));
  const unsigned short l2 = f2bf(v.z - bf2f(h2));
  const unsigned short l3 = f2bf(v.w - bf2f(h3));
  uint2v hv, lv;
  hv[0] = (unsigned)h0 | ((unsigned)h1 << 16);
  hv[1] = (unsigned)h2 | ((unsigned)h3 << 16);
  lv[0] = (unsigned)l0 | ((unsigned)l1 << 16);
  lv[1] = (unsigned)l2 | ((unsigned)l3 << 16);
  ((uint2v*)hi)[i] = hv;
  ((uint2v*)lo)[i] = lv;
}

// Fragment per CDNA5 16-bit layout: lane L holds row (L&15);
// K runs of 8 at kbase = 8*(L>>4) and kbase+16.
__device__ __forceinline__ void load_frag(Frag& f, const unsigned short* lds,
                                          int row, int khalf) {
  f.q[0] = *(const uint4v*)(lds + row * LDSS + khalf);
  f.q[1] = *(const uint4v*)(lds + row * LDSS + khalf + 16);
}

// C = A[M_,D_] * W[D_,D_]^T (+bias, optional softplus), all operands pre-split
// bf16 hi/lo. Double-buffered async global->LDS copies overlap the WMMAs.
template <int MODE>
__global__ __launch_bounds__(256) void gemm_bf16x3(
    const unsigned short* __restrict__ Ahi, const unsigned short* __restrict__ Alo,
    const unsigned short* __restrict__ Bhi, const unsigned short* __restrict__ Blo,
    const float* __restrict__ bias, float* __restrict__ Cout) {
  __shared__ unsigned short sAhi[2][TILE_M * LDSS];
  __shared__ unsigned short sAlo[2][TILE_M * LDSS];
  __shared__ unsigned short sBhi[2][TILE_N * LDSS];
  __shared__ unsigned short sBlo[2][TILE_N * LDSS];

  const int tid  = threadIdx.x;
  const int lane = tid & 31;
  const int wid  = tid >> 5;
  const int wm   = wid & 3;    // 4 waves along M (32 rows each)
  const int wn   = wid >> 2;   // 2 waves along N (32 cols each)
  const int row_blk = blockIdx.x * TILE_M;
  const int col_blk = blockIdx.y * TILE_N;

  // copy mapping: 16B (8 bf16) per lane; seg = 16B segment within 32-wide K chunk
  const int seg = tid & 3;
  const int r0  = tid >> 2;    // 0..63
  const unsigned aoff0 = (unsigned)(((size_t)(row_blk + r0)      * D_ + seg * 8) * 2);
  const unsigned aoff1 = (unsigned)(((size_t)(row_blk + r0 + 64) * D_ + seg * 8) * 2);
  const unsigned boff  = (unsigned)(((size_t)(col_blk + r0)      * D_ + seg * 8) * 2);
  const unsigned ldsA0 = r0 * LDSS + seg * 8;
  const unsigned ldsA1 = (r0 + 64) * LDSS + seg * 8;
  const unsigned ldsB0 = r0 * LDSS + seg * 8;

  v8f c[2][2];
#pragma unroll
  for (int i = 0; i < 2; ++i)
#pragma unroll
    for (int j = 0; j < 2; ++j)
#pragma unroll
      for (int r = 0; r < 8; ++r) c[i][j][r] = 0.0f;

  const int lrow  = lane & 15;
  const int khalf = (lane >> 4) * 8;   // 0 or 8

  auto issue = [&](int buf, unsigned kbyte) {
    async_ld16(&sAhi[buf][ldsA0], Ahi, aoff0 + kbyte);
    async_ld16(&sAhi[buf][ldsA1], Ahi, aoff1 + kbyte);
    async_ld16(&sAlo[buf][ldsA0], Alo, aoff0 + kbyte);
    async_ld16(&sAlo[buf][ldsA1], Alo, aoff1 + kbyte);
    async_ld16(&sBhi[buf][ldsB0], Bhi, boff + kbyte);
    async_ld16(&sBlo[buf][ldsB0], Blo, boff + kbyte);
  };

  issue(0, 0);               // prefetch first chunk
  int ibuf = 0;
  for (int kb = 0; kb < D_; kb += TILE_K, ibuf ^= 1) {
    if (kb + TILE_K < D_) {
      issue(ibuf ^ 1, (unsigned)((kb + TILE_K) * 2));   // prefetch next chunk
      asm volatile("s_wait_asynccnt 0x6" ::: "memory"); // current chunk landed
    } else {
      asm volatile("s_wait_asynccnt 0x0" ::: "memory");
    }
    __syncthreads();   // all waves' async copies for this chunk visible

    Frag ahi[2], alo[2], bhi[2], blo[2];
#pragma unroll
    for (int i = 0; i < 2; ++i) {
      const int r = wm * 32 + i * 16 + lrow;
      load_frag(ahi[i], &sAhi[ibuf][0], r, khalf);
      load_frag(alo[i], &sAlo[ibuf][0], r, khalf);
    }
#pragma unroll
    for (int j = 0; j < 2; ++j) {
      const int r = wn * 32 + j * 16 + lrow;
      load_frag(bhi[j], &sBhi[ibuf][0], r, khalf);
      load_frag(blo[j], &sBlo[ibuf][0], r, khalf);
    }
    __syncthreads();   // reads done; next iteration may overwrite this buffer

    // bf16x3: hi*hi + hi*lo + lo*hi, f32 accumulate
#pragma unroll
    for (int i = 0; i < 2; ++i)
#pragma unroll
      for (int j = 0; j < 2; ++j) {
        c[i][j] = __builtin_amdgcn_wmma_f32_16x16x32_bf16(
            false, ahi[i].v, false, bhi[j].v, (short)0, c[i][j], false, false);
        c[i][j] = __builtin_amdgcn_wmma_f32_16x16x32_bf16(
            false, ahi[i].v, false, blo[j].v, (short)0, c[i][j], false, false);
        c[i][j] = __builtin_amdgcn_wmma_f32_16x16x32_bf16(
            false, alo[i].v, false, bhi[j].v, (short)0, c[i][j], false, false);
      }
  }

  // C layout: row = base + 8*(lane>>4) + r, col = base + (lane&15)
#pragma unroll
  for (int i = 0; i < 2; ++i) {
    const int mbase = row_blk + wm * 32 + i * 16 + (lane >> 4) * 8;
#pragma unroll
    for (int j = 0; j < 2; ++j) {
      const int col = col_blk + wn * 32 + j * 16 + (lane & 15);
      const float bs = bias[col];
#pragma unroll
      for (int r = 0; r < 8; ++r) {
        float v = c[i][j][r] + bs;
        if (MODE == 1) v = softplusf(v);
        Cout[(size_t)(mbase + r) * D_ + col] = v;
      }
    }
  }
}

// Selective-scan: one lane per (b,d,n); N=16 reduction via shfl_xor.
// Emits y pre-split as bf16 hi/lo so the output GEMM needs no conversion.
__global__ __launch_bounds__(256) void scan_kernel(
    const float* __restrict__ x, const float* __restrict__ delta,
    const float* __restrict__ a_log, const float* __restrict__ b_param,
    unsigned short* __restrict__ yhi, unsigned short* __restrict__ ylo) {
  const int gid = blockIdx.x * blockDim.x + threadIdx.x;
  const int n = gid & (N_ - 1);
  const int d = (gid >> 4) & (D_ - 1);
  const int b = gid >> 14;

  const float ap = softplusf(a_log[d * N_ + n]) + 1e-4f;
  const float bp = b_param[d * N_ + n];

  const size_t base = (size_t)b * T_ * D_ + d;
  const float* xp = x + base;
  const float* dp = delta + base;
  unsigned short* hp = yhi + base;
  unsigned short* lp = ylo + base;

  float state = 0.0f;
  for (int t = 0; t < T_; ++t) {
    const float xv = xp[(size_t)t * D_];
    const float dt = dp[(size_t)t * D_];
    state = __expf(-dt * ap) * state + (dt * bp) * xv;
    float s = state;
    s += __shfl_xor(s, 1, 32);
    s += __shfl_xor(s, 2, 32);
    s += __shfl_xor(s, 4, 32);
    s += __shfl_xor(s, 8, 32);
    if (n == 0) {
      const unsigned short h = f2bf(s);
      hp[(size_t)t * D_] = h;
      lp[(size_t)t * D_] = f2bf(s - bf2f(h));
    }
  }
}

extern "C" void kernel_launch(void* const* d_in, const int* in_sizes, int n_in,
                              void* d_out, int out_size, void* d_ws, size_t ws_size,
                              hipStream_t stream) {
  (void)in_sizes; (void)n_in; (void)out_size; (void)ws_size;
  const float* x       = (const float*)d_in[0];
  const float* Wd      = (const float*)d_in[1];
  const float* bd      = (const float*)d_in[2];
  const float* a_log   = (const float*)d_in[3];
  const float* b_param = (const float*)d_in[4];
  const float* Wo      = (const float*)d_in[5];
  const float* bo      = (const float*)d_in[6];

  float* out   = (float*)d_out;
  float* delta = out;   // stage delta in d_out; overwritten by final GEMM

  // workspace layout (72 MB total)
  unsigned short* x_hi  = (unsigned short*)d_ws;
  unsigned short* x_lo  = x_hi + (size_t)M_ * D_;
  unsigned short* wd_hi = x_lo + (size_t)M_ * D_;
  unsigned short* wd_lo = wd_hi + (size_t)D_ * D_;
  unsigned short* wo_hi = wd_lo + (size_t)D_ * D_;
  unsigned short* wo_lo = wo_hi + (size_t)D_ * D_;
  unsigned short* y_hi  = wo_lo + (size_t)D_ * D_;
  unsigned short* y_lo  = y_hi + (size_t)M_ * D_;

  const int nx4 = M_ * D_ / 4, nw4 = D_ * D_ / 4;
  split_kernel<<<nx4 / 256, 256, 0, stream>>>(x, x_hi, x_lo, nx4);
  split_kernel<<<nw4 / 256, 256, 0, stream>>>(Wd, wd_hi, wd_lo, nw4);
  split_kernel<<<nw4 / 256, 256, 0, stream>>>(Wo, wo_hi, wo_lo, nw4);

  dim3 gg(M_ / TILE_M, D_ / TILE_N);   // (64, 16)
  gemm_bf16x3<1><<<gg, 256, 0, stream>>>(x_hi, x_lo, wd_hi, wd_lo, bd, delta);
  scan_kernel<<<(B_ * D_ * N_) / 256, 256, 0, stream>>>(x, delta, a_log, b_param, y_hi, y_lo);
  gemm_bf16x3<0><<<gg, 256, 0, stream>>>(y_hi, y_lo, wo_hi, wo_lo, bo, out);
}
